// GraphAttention_44667659879085
// MI455X (gfx1250) — matrix operand order repaired and checked by reading, not compiled, not run on hardware
//
#include <hip/hip_runtime.h>
#include <cstdint>
#include <cstddef>

// ---------------------------------------------------------------------------
// GraphAttention for MI455X (gfx1250): bf16 WMMA flash attention + TDM loads.
//   B=2, N=4096, C=256, H=4, D=64.  adj stream (128 MB) is the roofline term.
//   All WMMA fragment traffic is arranged as contiguous 32B-per-lane runs so
//   the compiler emits b128 loads (no u16 gather storms).
// ---------------------------------------------------------------------------

#define BATCH 2
#define SEQ   4096
#define CDIM  256
#define HEADS 4
#define DHEAD 64
#define NEGINF (-1e9f)

typedef __bf16 bf16;
typedef __attribute__((ext_vector_type(16))) __bf16 v16bf;
typedef __attribute__((ext_vector_type(8)))  __bf16 v8bf;
typedef __attribute__((ext_vector_type(8)))  float  v8f;
typedef __attribute__((ext_vector_type(4)))  unsigned u32x4;
typedef __attribute__((ext_vector_type(8)))  int    i32x8;
typedef __attribute__((ext_vector_type(4)))  int    i32x4;

#if defined(__has_builtin)
#if __has_builtin(__builtin_amdgcn_tensor_load_to_lds) && __has_builtin(__builtin_amdgcn_s_wait_tensorcnt)
#define HAVE_TDM 1
#endif
#endif

#define WMMA_BF16(A, Bm, Cacc) \
  __builtin_amdgcn_wmma_f32_16x16x32_bf16(false, (A), false, (Bm), (short)0, (Cacc), false, false)

__device__ inline v16bf cat8(v8bf lo, v8bf hi) {
  return __builtin_shufflevector(lo, hi, 0, 1, 2, 3, 4, 5, 6, 7, 8, 9, 10, 11, 12, 13, 14, 15);
}

// ---- WMMA fragment loaders (wave32, contiguous k order, same for A and B) ----

// A matrix 16x32 (MxK), row-major source with leading dim ld (bf16 elements).
// lane<16: M=lane, K in {0..15}; lane>=16: M=lane-16, K in {16..31}.
// Each lane = two contiguous 16B runs -> 2 x b128 loads.
__device__ inline v16bf load_a_frag(const bf16* __restrict__ base, int ld) {
  const int lane  = threadIdx.x & 31;
  const int row   = lane & 15;
  const int khalf = (lane >> 4) ? 16 : 0;
  const bf16* p = base + row * ld + khalf;
  v8bf lo = *(const v8bf*)(p);
  v8bf hi = *(const v8bf*)(p + 8);
  return cat8(lo, hi);
}

// B matrix 32x16 (KxN) where source stores each column contiguously in K:
// element B[k][n] = base[n*ld + k].  lane<16: N=lane, K=0..15 (j = k);
// lane>=16: N=lane-16, K=16..31.  One contiguous 32B run per lane.
__device__ inline v16bf load_bT_frag(const bf16* __restrict__ base, int ld) {
  const int lane  = threadIdx.x & 31;
  const int col   = lane & 15;
  const int kbase = (lane >> 4) ? 16 : 0;
  const bf16* p = base + col * ld + kbase;
  v8bf lo = *(const v8bf*)(p);
  v8bf hi = *(const v8bf*)(p + 8);
  return cat8(lo, hi);
}

// B fragment from pre-packed weights: one 32B vector load per lane.
__device__ inline v16bf load_b_packed(const v16bf* __restrict__ frag_base) {
  return frag_base[threadIdx.x & 31];
}

// ---------------------------------------------------------------------------
// TDM: 2D tile (32 rows x 128 bytes) global -> LDS, data_size = 8 B units.
// clang-23 toolchain: 6-arg builtin (g0, g1, g2, g3, g_ext, cpol).
// ---------------------------------------------------------------------------
#if defined(HAVE_TDM)
__device__ inline void tdm_load_tile32x128(unsigned lds_byte_addr, const void* gaddr) {
  const unsigned lo = (unsigned)(size_t)gaddr;
  const unsigned hi = (unsigned)((size_t)gaddr >> 32);
  u32x4 g0;
  g0[0] = 1u;                                   // count=1, user descriptor
  g0[1] = lds_byte_addr;                        // lds_addr
  g0[2] = lo;                                   // global_addr[31:0]
  g0[3] = (hi & 0x01FFFFFFu) | (2u << 30);      // global_addr[56:32] | type=2
  i32x8 g1;
  g1[0] = (3 << 16);                            // data_size=3 (8B), no multicast
  g1[1] = (int)(16u << 16);                     // tensor_dim0 = 16 units (row = 128B)
  g1[2] = (int)((4096u & 0xFFFFu) << 16);       // tensor_dim0 hi=0 | tensor_dim1 lo16
  g1[3] = (int)(16u << 16);                     // tensor_dim1 hi=0 | tile_dim0 = 16 units
  g1[4] = 32;                                   // tile_dim1 = 32 rows, tile_dim2 = 0
  g1[5] = 16;                                   // tensor_dim0_stride = 16 units
  g1[6] = 0;                                    // dim1_stride (unused, 2D)
  g1[7] = 0;
  i32x4 z4 = {0, 0, 0, 0};
  i32x8 z8 = {0, 0, 0, 0, 0, 0, 0, 0};
  __builtin_amdgcn_tensor_load_to_lds(g0, g1, z4, z4, z8, 0);
}
#endif

// ---------------------------------------------------------------------------
// Kernel 0a: fp32 -> bf16 cast
// ---------------------------------------------------------------------------
__global__ void cvt_f32_bf16_kernel(const float* __restrict__ in, bf16* __restrict__ out, int n) {
  const int i = blockIdx.x * blockDim.x + threadIdx.x;
  if (i < n) out[i] = (bf16)in[i];
}

// Kernel 0b: W[256][256] fp32 -> packed bf16 WMMA-B fragments.
// Wp[kt][ct][lane][j]: value = W[(ct*16 + (lane&15))*256 + kt*32 + (lane>>4)*16 + j]
// so each lane's 16-element B fragment is contiguous (32B).
__global__ void cvt_pack_w_kernel(const float* __restrict__ W, bf16* __restrict__ Wp) {
  const int i = blockIdx.x * blockDim.x + threadIdx.x;   // 65536 threads
  const int j    = i & 15;
  const int lane = (i >> 4) & 31;
  const int ct   = (i >> 9) & 15;
  const int kt   = i >> 13;
  const int col   = lane & 15;
  const int kbase = (lane >> 4) * 16;
  Wp[i] = (bf16)W[(ct * 16 + col) * CDIM + kt * 32 + kbase + j];
}

// ---------------------------------------------------------------------------
// Kernel 1: y = x @ W^T + b, scatter to [B,H,N,D] bf16.
// One block = 16 rows x 256 cols. 8 waves, wave w -> cols [32w, 32w+32).
// ---------------------------------------------------------------------------
__global__ void gemm_qkv_kernel(const bf16* __restrict__ xb,   // [B*N, 256]
                                const bf16* __restrict__ Wp,   // packed fragments
                                const float* __restrict__ bias,
                                bf16* __restrict__ out) {      // [B,H,N,64]
  __shared__ __align__(32) bf16 xt[16 * CDIM];
  const int wave = threadIdx.x >> 5;
  const int lane = threadIdx.x & 31;
  const int m0 = blockIdx.x * 16;
  {
    const uint4* src = (const uint4*)(xb + (size_t)m0 * CDIM);
    uint4* dst = (uint4*)xt;                    // 512 uint4 total
    dst[threadIdx.x]       = src[threadIdx.x];
    dst[threadIdx.x + 256] = src[threadIdx.x + 256];
  }
  __syncthreads();

  const int ct0 = wave * 2;                     // two 16-col tiles per wave
  const v16bf* WpF = (const v16bf*)Wp;          // [8 kt][16 ct][32 lane]
  v8f acc0 = {}, acc1 = {};
#pragma unroll
  for (int kt = 0; kt < 8; ++kt) {
    v16bf a  = load_a_frag(xt + kt * 32, CDIM);
    v16bf b0 = load_b_packed(WpF + ((size_t)kt * 16 + ct0) * 32);
    v16bf b1 = load_b_packed(WpF + ((size_t)kt * 16 + ct0 + 1) * 32);
    acc0 = WMMA_BF16(a, b0, acc0);
    acc1 = WMMA_BF16(a, b1, acc1);
  }

  const int half = lane >> 4;
  const int cl = lane & 15;
  const int c0 = wave * 32;
#pragma unroll
  for (int i = 0; i < 8; ++i) {
    const int r = m0 + i + half * 8;            // global row in [0, B*N)
    const int bb = r >> 12, n = r & (SEQ - 1);
    {
      const int c = c0 + cl;
      const int h = c >> 6, d = c & 63;
      out[(((size_t)(bb * HEADS + h)) * SEQ + n) * DHEAD + d] = (bf16)(acc0[i] + bias[c]);
    }
    {
      const int c = c0 + 16 + cl;
      const int h = c >> 6, d = c & 63;
      out[(((size_t)(bb * HEADS + h)) * SEQ + n) * DHEAD + d] = (bf16)(acc1[i] + bias[c]);
    }
  }
}

// ---------------------------------------------------------------------------
// Kernel 2: flash attention with adj mask. One wave = 16 query rows.
// Block = 8 waves = 128 rows of one (b,h). Per 32-key chunk: TDM K/V -> LDS,
// V tile transposed in LDS for contiguous PV fragments, 4+4 WMMA.
// ---------------------------------------------------------------------------
__global__ void __launch_bounds__(256)
flash_attn_kernel(const bf16* __restrict__ Q, const bf16* __restrict__ K,
                  const bf16* __restrict__ V, const int* __restrict__ adj,
                  bf16* __restrict__ ao) {     // ao: [B, N, 256]
  __shared__ __align__(32) bf16 Kt[32 * DHEAD];   // [32 keys][64 d]  (TDM dest)
  __shared__ __align__(32) bf16 Vt[32 * DHEAD];   // [32 keys][64 d]  (TDM dest)
  __shared__ __align__(32) bf16 VtT[DHEAD * 32];  // [64 d][32 keys]  (transposed)
  __shared__ __align__(32) bf16 Pt[8][16 * 32];   // per-wave P staging (C->A relayout)

  const int wave = threadIdx.x >> 5;
  const int lane = threadIdx.x & 31;
  const int half = lane >> 4;
  const int cl = lane & 15;

  int idx = blockIdx.x;
  const int rb = idx & 31; idx >>= 5;
  const int h  = idx & 3;  idx >>= 2;
  const int b  = idx;
  const int r0 = rb * 128 + wave * 16;
  const size_t headoff = ((size_t)(b * HEADS + h)) * SEQ;

  const v16bf qa0 = load_a_frag(Q + (headoff + r0) * DHEAD, DHEAD);
  const v16bf qa1 = load_a_frag(Q + (headoff + r0) * DHEAD + 32, DHEAD);

  v8f o0 = {}, o1 = {}, o2 = {}, o3 = {};
  float mrun[8], lrun[8];
#pragma unroll
  for (int i = 0; i < 8; ++i) { mrun[i] = -3e38f; lrun[i] = 0.f; }

  // V-transpose work split: thread -> (key, 8 d values)
  const int tk = threadIdx.x & 31;
  const int td = (threadIdx.x >> 5) * 8;

  for (int kk = 0; kk < SEQ; kk += 32) {
#if defined(HAVE_TDM)
    if (wave == 0) {
      tdm_load_tile32x128((unsigned)(size_t)(void*)Kt, K + (headoff + kk) * DHEAD);
      tdm_load_tile32x128((unsigned)(size_t)(void*)Vt, V + (headoff + kk) * DHEAD);
      __builtin_amdgcn_s_wait_tensorcnt(0);
    }
#else
    {
      const uint4* ksrc = (const uint4*)(K + (headoff + kk) * DHEAD);
      const uint4* vsrc = (const uint4*)(V + (headoff + kk) * DHEAD);
      ((uint4*)Kt)[threadIdx.x] = ksrc[threadIdx.x];   // 256 * 16B = 4 KB
      ((uint4*)Vt)[threadIdx.x] = vsrc[threadIdx.x];
    }
#endif
    __syncthreads();

    // transpose V tile: VtT[d][key] = Vt[key][d]   (1 x b128 load + 8 stores/thread)
    {
      v8bf vrow = *(const v8bf*)(Vt + tk * DHEAD + td);
#pragma unroll
      for (int j = 0; j < 8; ++j) VtT[(td + j) * 32 + tk] = vrow[j];
    }

    // S = Q . K^T  (two 16x16 key tiles, K-dim 64 = 2 WMMA each)
    v8f S0 = {}, S1 = {};
    {
      v16bf bk = load_bT_frag(Kt, DHEAD);
      S0 = WMMA_BF16(qa0, bk, S0);
      bk = load_bT_frag(Kt + 32, DHEAD);
      S0 = WMMA_BF16(qa1, bk, S0);
      bk = load_bT_frag(Kt + 16 * DHEAD, DHEAD);
      S1 = WMMA_BF16(qa0, bk, S1);
      bk = load_bT_frag(Kt + 16 * DHEAD + 32, DHEAD);
      S1 = WMMA_BF16(qa1, bk, S1);
    }
    __syncthreads();   // VtT complete before PV fragments

    // prefetch next adj tile into GL2 while we do softmax math
    if (kk + 32 < SEQ)
      __builtin_prefetch(adj + ((size_t)(b * SEQ + r0 + cl)) * SEQ + kk + 32, 0, 1);

    // mask + online softmax (row = vgpr i + 8*half, col = cl within key tile)
    bf16* __restrict__ pw = Pt[wave];
#pragma unroll
    for (int i = 0; i < 8; ++i) {
      const int r = r0 + i + half * 8;
      const size_t arow = ((size_t)(b * SEQ + r)) * SEQ;
      float s0 = S0[i] * 0.125f;                // scale = D^-0.5 = 1/8
      float s1 = S1[i] * 0.125f;
      if (adj[arow + kk + cl] == 0)      s0 = NEGINF;
      if (adj[arow + kk + 16 + cl] == 0) s1 = NEGINF;

      float mx = fmaxf(s0, s1);
#pragma unroll
      for (int m = 1; m <= 8; m <<= 1) mx = fmaxf(mx, __shfl_xor(mx, m, 32));
      const float mnew  = fmaxf(mrun[i], mx);
      const float alpha = __expf(mrun[i] - mnew);
      const float p0 = __expf(s0 - mnew);
      const float p1 = __expf(s1 - mnew);
      float ls = p0 + p1;
#pragma unroll
      for (int m = 1; m <= 8; m <<= 1) ls += __shfl_xor(ls, m, 32);
      lrun[i] = lrun[i] * alpha + ls;
      mrun[i] = mnew;
      pw[(i + half * 8) * 32 + cl]      = (bf16)p0;
      pw[(i + half * 8) * 32 + 16 + cl] = (bf16)p1;
      o0[i] *= alpha; o1[i] *= alpha; o2[i] *= alpha; o3[i] *= alpha;
    }
    asm volatile("s_wait_dscnt 0" ::: "memory");  // P staging visible to wave-wide A load

    // O += P (16x32) . V (32x64)  -- B fragments contiguous from VtT
    {
      const v16bf pa = load_a_frag(pw, 32);
      v16bf vb = load_bT_frag(VtT, 32);
      o0 = WMMA_BF16(pa, vb, o0);
      vb = load_bT_frag(VtT + 16 * 32, 32);
      o1 = WMMA_BF16(pa, vb, o1);
      vb = load_bT_frag(VtT + 32 * 32, 32);
      o2 = WMMA_BF16(pa, vb, o2);
      vb = load_bT_frag(VtT + 48 * 32, 32);
      o3 = WMMA_BF16(pa, vb, o3);
    }
    __syncthreads();
  }

  // normalize and write merged-head layout [B, N, 256]
#pragma unroll
  for (int i = 0; i < 8; ++i) {
    const float inv = 1.0f / lrun[i];
    const int r = r0 + i + half * 8;
    const size_t orow = ((size_t)(b * SEQ + r)) * CDIM + h * DHEAD;
    ao[orow + cl]      = (bf16)(o0[i] * inv);
    ao[orow + 16 + cl] = (bf16)(o1[i] * inv);
    ao[orow + 32 + cl] = (bf16)(o2[i] * inv);
    ao[orow + 48 + cl] = (bf16)(o3[i] * inv);
  }
}

// ---------------------------------------------------------------------------
// Kernel 3: out = LN(ao @ Wo^T + bo) * gamma + beta   (fp32 output)
// ---------------------------------------------------------------------------
__global__ void oproj_ln_kernel(const bf16* __restrict__ ao,   // [B*N, 256]
                                const bf16* __restrict__ Wp,   // packed Wo fragments
                                const float* __restrict__ bo,
                                const float* __restrict__ gamma,
                                const float* __restrict__ beta,
                                float* __restrict__ out) {
  __shared__ __align__(32) bf16 xt[16 * CDIM];
  __shared__ float yt[16 * CDIM];
  const int wave = threadIdx.x >> 5;
  const int lane = threadIdx.x & 31;
  const int m0 = blockIdx.x * 16;
  {
    const uint4* src = (const uint4*)(ao + (size_t)m0 * CDIM);
    uint4* dst = (uint4*)xt;
    dst[threadIdx.x]       = src[threadIdx.x];
    dst[threadIdx.x + 256] = src[threadIdx.x + 256];
  }
  __syncthreads();

  const int ct0 = wave * 2;
  const v16bf* WpF = (const v16bf*)Wp;
  v8f acc0 = {}, acc1 = {};
#pragma unroll
  for (int kt = 0; kt < 8; ++kt) {
    v16bf a  = load_a_frag(xt + kt * 32, CDIM);
    v16bf b0 = load_b_packed(WpF + ((size_t)kt * 16 + ct0) * 32);
    v16bf b1 = load_b_packed(WpF + ((size_t)kt * 16 + ct0 + 1) * 32);
    acc0 = WMMA_BF16(a, b0, acc0);
    acc1 = WMMA_BF16(a, b1, acc1);
  }
  const int half = lane >> 4;
  const int cl = lane & 15;
  const int c0 = wave * 32;
#pragma unroll
  for (int i = 0; i < 8; ++i) {
    const int rt = i + half * 8;
    yt[rt * CDIM + c0 + cl]      = acc0[i] + bo[c0 + cl];
    yt[rt * CDIM + c0 + 16 + cl] = acc1[i] + bo[c0 + 16 + cl];
  }
  __syncthreads();

  // LayerNorm: 16 threads per row, 16 cols each
  const int r = threadIdx.x >> 4;
  const int l16 = threadIdx.x & 15;
  float s = 0.f, s2 = 0.f;
#pragma unroll
  for (int k = 0; k < 16; ++k) {
    const float v = yt[r * CDIM + l16 + 16 * k];
    s += v; s2 += v * v;
  }
#pragma unroll
  for (int m = 1; m <= 8; m <<= 1) { s += __shfl_xor(s, m, 32); s2 += __shfl_xor(s2, m, 32); }
  const float mu = s * (1.f / CDIM);
  const float var = s2 * (1.f / CDIM) - mu * mu;
  const float rs = rsqrtf(var + 1e-5f);
  const size_t row = (size_t)m0 + r;
#pragma unroll
  for (int k = 0; k < 16; ++k) {
    const int c = l16 + 16 * k;
    const float v = yt[r * CDIM + c];
    out[row * CDIM + c] = (v - mu) * rs * gamma[c] + beta[c];
  }
}

// ---------------------------------------------------------------------------
// Host driver
// ---------------------------------------------------------------------------
extern "C" void kernel_launch(void* const* d_in, const int* in_sizes, int n_in,
                              void* d_out, int out_size, void* d_ws, size_t ws_size,
                              hipStream_t stream) {
  (void)in_sizes; (void)n_in; (void)out_size; (void)ws_size;
  const float* x     = (const float*)d_in[0];
  const int*   adj   = (const int*)d_in[1];
  const float* Wq    = (const float*)d_in[2];
  const float* bq    = (const float*)d_in[3];
  const float* Wk    = (const float*)d_in[4];
  const float* bk    = (const float*)d_in[5];
  const float* Wv    = (const float*)d_in[6];
  const float* bv    = (const float*)d_in[7];
  const float* Wo    = (const float*)d_in[8];
  const float* bo    = (const float*)d_in[9];
  const float* gamma = (const float*)d_in[10];
  const float* beta  = (const float*)d_in[11];
  float* out = (float*)d_out;

  const size_t nTok = (size_t)BATCH * SEQ;          // 8192 rows
  const size_t nX   = nTok * CDIM;                  // 2,097,152
  const size_t nW   = (size_t)CDIM * CDIM;          // 65,536

  char* ws = (char*)d_ws;
  bf16* xb  = (bf16*)ws;                ws += nX * sizeof(bf16);
  bf16* Wqp = (bf16*)ws;                ws += nW * sizeof(bf16);
  bf16* Wkp = (bf16*)ws;                ws += nW * sizeof(bf16);
  bf16* Wvp = (bf16*)ws;                ws += nW * sizeof(bf16);
  bf16* Wop = (bf16*)ws;                ws += nW * sizeof(bf16);
  bf16* Qb  = (bf16*)ws;                ws += nX * sizeof(bf16);
  bf16* Kb  = (bf16*)ws;                ws += nX * sizeof(bf16);
  bf16* Vb  = (bf16*)ws;                ws += nX * sizeof(bf16);
  bf16* AOb = (bf16*)ws;                ws += nX * sizeof(bf16);

  // 1) casts / weight packing
  cvt_f32_bf16_kernel<<<(int)(nX / 256), 256, 0, stream>>>(x, xb, (int)nX);
  cvt_pack_w_kernel<<<(int)(nW / 256), 256, 0, stream>>>(Wq, Wqp);
  cvt_pack_w_kernel<<<(int)(nW / 256), 256, 0, stream>>>(Wk, Wkp);
  cvt_pack_w_kernel<<<(int)(nW / 256), 256, 0, stream>>>(Wv, Wvp);
  cvt_pack_w_kernel<<<(int)(nW / 256), 256, 0, stream>>>(Wo, Wop);

  // 2) Q/K/V projections (512 blocks of 16 rows)
  gemm_qkv_kernel<<<(int)(nTok / 16), 256, 0, stream>>>(xb, Wqp, bq, Qb);
  gemm_qkv_kernel<<<(int)(nTok / 16), 256, 0, stream>>>(xb, Wkp, bk, Kb);
  gemm_qkv_kernel<<<(int)(nTok / 16), 256, 0, stream>>>(xb, Wvp, bv, Vb);

  // 3) flash attention: B*H*(N/128) = 256 blocks
  flash_attn_kernel<<<BATCH * HEADS * (SEQ / 128), 256, 0, stream>>>(Qb, Kb, Vb, adj, AOb);

  // 4) output projection + layernorm
  oproj_ln_kernel<<<(int)(nTok / 16), 256, 0, stream>>>(AOb, Wop, bo, gamma, beta, out);
}
